// AMS_73349451481283
// MI455X (gfx1250) — compile-verified, matrix operand order checked by reading, not run on hardware
//
#include <hip/hip_runtime.h>
#include <hip/hip_fp16.h>
#include <math.h>

// ---------------- problem constants (match reference) ----------------
#define Bn   32
#define Ln   192
#define Nn   64
#define Dn   32
#define En   4
#define DFFn 64
#define Hn   4
#define OUTMAIN (Bn*Ln*Nn*Dn)   // 12582912
#define PI_F 3.14159265358979f

typedef __attribute__((ext_vector_type(16))) _Float16 v16h;
typedef __attribute__((ext_vector_type(8)))  float    v8f;

union UF16 { v16h v; _Float16 e[16]; uint4 q[2]; };
union UF8  { v8f  v; float     e[8]; };
union PK4  { _Float16 h[4]; uint2 u; };

__device__ __forceinline__ float nan0f(float v) {
    return (__builtin_isnan(v) || __builtin_isinf(v)) ? 0.0f : v;
}
__device__ __forceinline__ float gelu_f(float x) {
    float x3 = x * x * x;
    return 0.5f * x * (1.0f + tanhf(0.7978845608028654f * (x + 0.044715f * x3)));
}
__device__ __forceinline__ float cv_sq4(const float* v) {
    float m = 0.f;
    for (int e = 0; e < 4; e++) m += v[e];
    m *= 0.25f;
    float var = 0.f;
    for (int e = 0; e < 4; e++) { float u = v[e] - m; var += u * u; }
    var *= (1.0f / 3.0f);                       // ddof=1
    return var / (m * m + 1e-10f);
}
// fragment = two contiguous 8-half (16B) runs in LDS -> two ds_load_b128
__device__ __forceinline__ void ldfrag(UF16& f, const _Float16* p) {
    f.q[0] = *(const uint4*)p;
    f.q[1] = *(const uint4*)(p + 16);
}

// ---------------- stage 0: out = x ----------------
__global__ void k_init_out(const float4* __restrict__ x, float4* __restrict__ out) {
    int i = blockIdx.x * 256 + threadIdx.x;
    if (i < OUTMAIN / 4) out[i] = x[i];
}

// ---------------- stage 1: trend + season -> new_x ----------------
// one block per (b,n), 192 threads (one per time step l)
__global__ void k_season_trend(const float* __restrict__ x,
                               const float* __restrict__ w_dec, const float* __restrict__ b_dec,
                               float* __restrict__ newx) {
    const int b = blockIdx.x / Nn, n = blockIdx.x % Nn;
    const int l = threadIdx.x;
    __shared__ float xs[Ln];
    __shared__ float re[96], im[96], mag[96];
    __shared__ int topi[3];
    xs[l] = x[((b * Ln + l) * Nn + n) * Dn + 0];
    __syncthreads();

    // trend: weighted moving averages (kernels 4,8,12), edge-replicated
    float tr = 0.f;
    {
        const int ks[3] = {4, 8, 12};
        float ma[3];
        for (int j = 0; j < 3; j++) {
            int k = ks[j], pf = (k - 1) / 2;
            float s = 0.f;
            for (int i = 0; i < k; i++) {
                int t = l - pf + i;
                t = t < 0 ? 0 : (t >= Ln ? Ln - 1 : t);
                s += xs[t];
            }
            ma[j] = s / (float)k;
        }
        float xv = xs[l], w[3], mx = -1e30f;
        for (int j = 0; j < 3; j++) { w[j] = xv * w_dec[j] + b_dec[j]; mx = fmaxf(mx, w[j]); }
        float sm = 0.f;
        for (int j = 0; j < 3; j++) { w[j] = expf(w[j] - mx); sm += w[j]; }
        for (int j = 0; j < 3; j++) tr += ma[j] * (w[j] / sm);
    }

    // rDFT bins 1..95 (exclude DC and Nyquist), threads 1..95
    if (l >= 1 && l <= 95) {
        float rr = 0.f, ii = 0.f;
        float w0 = -2.0f * PI_F * (float)l / (float)Ln;
        for (int t = 0; t < Ln; t++) {
            float s, c;
            sincosf(w0 * (float)t, &s, &c);
            rr += xs[t] * c;
            ii += xs[t] * s;
        }
        re[l] = rr; im[l] = ii; mag[l] = sqrtf(rr * rr + ii * ii);
    }
    __syncthreads();
    if (l == 0) {
        float m1 = -1.f; int i1 = 1;
        for (int f = 1; f <= 95; f++) if (mag[f] > m1) { m1 = mag[f]; i1 = f; }
        float m2 = -1.f; int i2 = (i1 == 1) ? 2 : 1;
        for (int f = 1; f <= 95; f++) if (f != i1 && mag[f] > m2) { m2 = mag[f]; i2 = f; }
        float m3 = -1.f; int i3 = 1;
        for (int f = 1; f <= 95; f++) if (f != i1 && f != i2 && mag[f] > m3) { m3 = mag[f]; i3 = f; }
        topi[0] = i1; topi[1] = i2; topi[2] = i3;
    }
    __syncthreads();
    float season = 0.f;
    for (int k = 0; k < 3; k++) {
        int f = topi[k];
        float amp = mag[f] / (float)Ln;
        float ph  = atan2f(im[f], re[f]);
        season += 2.0f * amp * cosf(2.0f * PI_F * ((float)f / (float)Ln) * (float)l + ph);
    }
    newx[(b * Ln + l) * Nn + n] = nan0f(xs[l] + season + tr);
}

// g[b,l] = sum_n new_x * w_start + b_start
__global__ void k_g(const float* __restrict__ newx, const float* __restrict__ w_start,
                    const float* __restrict__ b_start, float* __restrict__ g) {
    int i = blockIdx.x * 256 + threadIdx.x;
    if (i >= Bn * Ln) return;
    const float* row = &newx[i * Nn];
    float s = 0.f;
    for (int n = 0; n < Nn; n++) s += row[n] * w_start[n];
    g[i] = nan0f(s + b_start[0]);
}

// gating: logits, top-2 softmax, balance loss
__global__ void k_gate(const float* __restrict__ g, const float* __restrict__ w_gate,
                       const float* __restrict__ b_gate, float* __restrict__ gates,
                       float* __restrict__ loss_out) {
    __shared__ float gsh[Bn * En];
    const int b = threadIdx.x;
    if (b < Bn) {
        float lg[En];
        for (int e = 0; e < En; e++) {
            float s = b_gate[e];
            for (int l = 0; l < Ln; l++) s += g[b * Ln + l] * w_gate[l * En + e];
            lg[e] = nan0f(s);
        }
        int i1 = 0;
        for (int e = 1; e < En; e++) if (lg[e] > lg[i1]) i1 = e;
        int i2 = -1;
        for (int e = 0; e < En; e++) { if (e == i1) continue; if (i2 < 0 || lg[e] > lg[i2]) i2 = e; }
        float m = fmaxf(lg[i1], lg[i2]);
        float e1 = expf(lg[i1] - m), e2 = expf(lg[i2] - m);
        float inv = 1.0f / (e1 + e2);
        for (int e = 0; e < En; e++) gsh[b * En + e] = 0.f;
        gsh[b * En + i1] = e1 * inv;
        gsh[b * En + i2] = e2 * inv;
        for (int e = 0; e < En; e++) gates[b * En + e] = gsh[b * En + e];
    }
    __syncthreads();
    if (threadIdx.x == 0) {
        float imp[En], ld[En];
        for (int e = 0; e < En; e++) {
            float si = 0.f, sl = 0.f;
            for (int bb = 0; bb < Bn; bb++) {
                float v = gsh[bb * En + e];
                si += v;
                if (v > 0.f) sl += 1.f;
            }
            imp[e] = si; ld[e] = sl;
        }
        loss_out[0] = (cv_sq4(imp) + cv_sq4(ld)) * 0.01f;
    }
}

// ---------------- expert kernels ----------------
// pack z0[s,Pi,k] = x[b, Pi*p + t, n, d] with k = d*p + t
template <int p>
__global__ void k_pack(const float* __restrict__ x, float* __restrict__ z0) {
    constexpr int P = Ln / p, Dp = Dn * p;
    int i = blockIdx.x * 256 + threadIdx.x;
    if (i >= OUTMAIN) return;
    int k  = i % Dp;
    int Pi = (i / Dp) % P;
    int s  = i / (Dp * P);
    int b = s / Nn, n = s % Nn;
    int d = k / p, t = k % p;
    z0[i] = x[((b * Ln + Pi * p + t) * Nn + n) * Dn + d];
}

// WMMA tiled GEMM: C[R,Nc] = A[R,K] @ Bw[K,Nc] (+bias[col]) (+wpos[(row%Pmod)*Nc+col])
// 64x64 block tile, 128 threads = 4 waves; each wave owns 32x32 (2x2 WMMA tiles).
// Double-buffered LDS (ping-pong): global loads for tile i+1 are issued before the
// WMMAs of tile i, convert+ds_store after, one barrier per K-step.
// A staged row-major (stride 32 halfs), B staged transposed (Bs[col][k]) so both
// fragments are two contiguous 16B runs per lane -> 2x ds_load_b128 each.
__global__ void __launch_bounds__(128) k_gemm(
        const float* __restrict__ A, const float* __restrict__ Bw,
        const float* __restrict__ bias, const float* __restrict__ wpos,
        float* __restrict__ C, int R, int K, int Nc, int Pmod) {
    __shared__ _Float16 As[2][64 * 32];   // As[buf][r*32 + k]
    __shared__ _Float16 Bs[2][64 * 32];   // Bs[buf][c*32 + k]  (transposed)
    const int tid  = threadIdx.x;
    const int wid  = tid >> 5;
    const int lane = tid & 31;
    const int row0 = blockIdx.x * 64;
    const int col0 = blockIdx.y * 64;
    const int wr = (wid >> 1) * 32;    // wave row offset in block tile
    const int wc = (wid & 1) * 32;     // wave col offset in block tile

    const int m  = lane & 15;
    const int kb = (lane >= 16) ? 8 : 0;
    const int ar = tid >> 3;           // 0..15, A stage row base
    const int ac = (tid & 7) * 4;      // A stage col (multiple of 4)
    const int bc = tid & 63;           // B stage column
    const int bg = tid >> 6;           // 0/1, B stage row-group base

    UF8 acc[2][2];
#pragma unroll
    for (int i = 0; i < 2; i++)
#pragma unroll
        for (int j = 0; j < 2; j++)
#pragma unroll
            for (int r = 0; r < 8; r++) acc[i][j].e[r] = 0.f;

    float4 va[4];            // in-flight A tile (registers)
    float  vb[4][4];         // in-flight B tile (registers)

    auto load_tiles = [&](int kk) {
#pragma unroll
        for (int i = 0; i < 4; i++)
            va[i] = *(const float4*)&A[(size_t)(row0 + ar + 16 * i) * K + kk + ac];
#pragma unroll
        for (int i = 0; i < 4; i++) {
            int r0 = (bg + 2 * i) * 4;
#pragma unroll
            for (int j = 0; j < 4; j++)
                vb[i][j] = Bw[(size_t)(kk + r0 + j) * Nc + col0 + bc];
        }
    };
    auto store_tiles = [&](int buf) {
#pragma unroll
        for (int i = 0; i < 4; i++) {
            PK4 pk;
            pk.h[0] = (_Float16)va[i].x; pk.h[1] = (_Float16)va[i].y;
            pk.h[2] = (_Float16)va[i].z; pk.h[3] = (_Float16)va[i].w;
            *(uint2*)&As[buf][(ar + 16 * i) * 32 + ac] = pk.u;
        }
#pragma unroll
        for (int i = 0; i < 4; i++) {
            int r0 = (bg + 2 * i) * 4;
            PK4 pk;
            pk.h[0] = (_Float16)vb[i][0]; pk.h[1] = (_Float16)vb[i][1];
            pk.h[2] = (_Float16)vb[i][2]; pk.h[3] = (_Float16)vb[i][3];
            *(uint2*)&Bs[buf][bc * 32 + r0] = pk.u;
        }
    };

    // prologue: stage tile 0
    load_tiles(0);
    store_tiles(0);
    __syncthreads();

    int buf = 0;
    for (int kk = 0; kk < K; kk += 32) {
        const bool more = (kk + 32 < K);
        if (more) load_tiles(kk + 32);            // global loads fly during WMMAs

        UF16 fa0, fa1, fb0, fb1;
        ldfrag(fa0, &As[buf][(wr + m) * 32 + kb]);
        ldfrag(fa1, &As[buf][(wr + 16 + m) * 32 + kb]);
        ldfrag(fb0, &Bs[buf][(wc + m) * 32 + kb]);
        ldfrag(fb1, &Bs[buf][(wc + 16 + m) * 32 + kb]);
        acc[0][0].v = __builtin_amdgcn_wmma_f32_16x16x32_f16(false, fa0.v, false, fb0.v,
                                                             (short)0, acc[0][0].v, false, false);
        acc[0][1].v = __builtin_amdgcn_wmma_f32_16x16x32_f16(false, fa0.v, false, fb1.v,
                                                             (short)0, acc[0][1].v, false, false);
        acc[1][0].v = __builtin_amdgcn_wmma_f32_16x16x32_f16(false, fa1.v, false, fb0.v,
                                                             (short)0, acc[1][0].v, false, false);
        acc[1][1].v = __builtin_amdgcn_wmma_f32_16x16x32_f16(false, fa1.v, false, fb1.v,
                                                             (short)0, acc[1][1].v, false, false);

        if (more) store_tiles(buf ^ 1);           // convert + ds_store into other buffer
        __syncthreads();
        buf ^= 1;
    }

    const int nlo = lane & 15;
    const int mh  = (lane >= 16) ? 8 : 0;
#pragma unroll
    for (int ti = 0; ti < 2; ti++)
#pragma unroll
        for (int tj = 0; tj < 2; tj++) {
#pragma unroll
            for (int r = 0; r < 8; r++) {
                int row = row0 + wr + ti * 16 + r + mh;
                int col = col0 + wc + tj * 16 + nlo;
                float v = acc[ti][tj].e[r];
                if (bias) v += bias[col];
                if (wpos) v += wpos[(row % Pmod) * Nc + col];
                C[(size_t)row * Nc + col] = v;
            }
        }
}

// inter attention: one block per (s, h, row); softmax over P keys
template <int p>
__global__ void k_attn(const float* __restrict__ Q, const float* __restrict__ Km,
                       const float* __restrict__ V, float* __restrict__ O) {
    constexpr int P = Ln / p, Dp = Dn * p, dh = Dp / Hn;
    const int gb = blockIdx.x;
    const int Pr = gb % P;
    const int h  = (gb / P) % Hn;
    const int s  = gb / (P * Hn);
    __shared__ float qv[dh];
    __shared__ float sc[P];
    __shared__ float rsum;
    const float scale = rsqrtf((float)dh);
    const size_t base = (size_t)s * P * Dp + (size_t)h * dh;

    for (int j = threadIdx.x; j < dh; j += blockDim.x) qv[j] = Q[base + (size_t)Pr * Dp + j];
    __syncthreads();
    for (int t = threadIdx.x; t < P; t += blockDim.x) {
        const float* kr = &Km[base + (size_t)t * Dp];
        float d = 0.f;
        for (int j = 0; j < dh; j++) d += qv[j] * kr[j];
        sc[t] = d * scale;
    }
    __syncthreads();
    if (threadIdx.x == 0) {
        float mx = -1e30f;
        for (int t = 0; t < P; t++) mx = fmaxf(mx, sc[t]);
        float sm = 0.f;
        for (int t = 0; t < P; t++) { float e = expf(sc[t] - mx); sc[t] = e; sm += e; }
        rsum = 1.0f / sm;
    }
    __syncthreads();
    const float rs = rsum;
    for (int j = threadIdx.x; j < dh; j += blockDim.x) {
        float o = 0.f;
        for (int t = 0; t < P; t++) o += sc[t] * V[base + (size_t)t * Dp + j];
        O[base + (size_t)Pr * Dp + j] = o * rs;
    }
}

// batch-independent intra-attention precompute: qh = (emb@W+b)@wq etc.
__global__ void k_intra_q(const float* __restrict__ intra_emb, const float* __restrict__ emb_w,
                          const float* __restrict__ emb_b,
                          const float* __restrict__ wq, const float* __restrict__ wk,
                          const float* __restrict__ wv,
                          float* __restrict__ qh_c, float* __restrict__ kh0,
                          float* __restrict__ vh0, int P) {
    int idx = blockIdx.x * 128 + threadIdx.x;     // over P*N
    if (idx >= P * Nn) return;
    int Pi = idx / Nn, n = idx % Nn;
    float qc[Dn];
    for (int d = 0; d < Dn; d++) {
        float s = emb_b[Pi * Dn + d];
        for (int m = 0; m < 16; m++)
            s += intra_emb[(Pi * Nn + n) * 16 + m] * emb_w[(Pi * 16 + m) * Dn + d];
        qc[d] = s;
    }
    for (int d = 0; d < Dn; d++) {
        float sq = 0.f, sk = 0.f, sv = 0.f;
        for (int dd = 0; dd < Dn; dd++) {
            float q = qc[dd];
            sq += q * wq[dd * Dn + d];
            sk += q * wk[dd * Dn + d];
            sv += q * wv[dd * Dn + d];
        }
        qh_c[idx * Dn + d] = sq;
        kh0[idx * Dn + d]  = sk;
        vh0[idx * Dn + d]  = sv;
    }
}

// intra attention: one thread per (b, Pi, n); kv = [q, p patch rows]
template <int p>
__global__ void k_intra(const float* __restrict__ x, const float* __restrict__ wk,
                        const float* __restrict__ wv,
                        const float* __restrict__ qh_c, const float* __restrict__ kh0,
                        const float* __restrict__ vh0, float* __restrict__ ipre) {
    constexpr int P = Ln / p;
    __shared__ float swk[Dn * Dn], swv[Dn * Dn];
    for (int i = threadIdx.x; i < Dn * Dn; i += blockDim.x) { swk[i] = wk[i]; swv[i] = wv[i]; }
    __syncthreads();
    int idx = blockIdx.x * blockDim.x + threadIdx.x;     // over B*P*N
    if (idx >= Bn * P * Nn) return;
    const int n  = idx % Nn;
    const int Pi = (idx / Nn) % P;
    const int b  = idx / (Nn * P);
    const float* qh = &qh_c[(Pi * Nn + n) * Dn];
    float wkq[Dn];                                      // wk @ qh
    for (int dd = 0; dd < Dn; dd++) {
        float s = 0.f;
        for (int d = 0; d < Dn; d++) s += swk[dd * Dn + d] * qh[d];
        wkq[dd] = s;
    }
    float sc[p + 1];
    {
        const float* k0 = &kh0[(Pi * Nn + n) * Dn];
        float s = 0.f;
        for (int d = 0; d < Dn; d++) s += qh[d] * k0[d];
        sc[0] = s;
    }
    for (int t = 1; t <= p; t++) {
        const float* xr = &x[((size_t)(b * Ln + Pi * p + (t - 1)) * Nn + n) * Dn];
        float s = 0.f;
        for (int dd = 0; dd < Dn; dd++) s += xr[dd] * wkq[dd];
        sc[t] = s;
    }
    const float inv = 0.1767766952966369f;   // 1/sqrt(32)
    float mx = -1e30f;
    for (int t = 0; t <= p; t++) { sc[t] *= inv; mx = fmaxf(mx, sc[t]); }
    float sm = 0.f;
    for (int t = 0; t <= p; t++) { sc[t] = expf(sc[t] - mx); sm += sc[t]; }
    const float rs = 1.0f / sm;
    float xacc[Dn];
    for (int d = 0; d < Dn; d++) xacc[d] = 0.f;
    for (int t = 1; t <= p; t++) {
        const float* xr = &x[((size_t)(b * Ln + Pi * p + (t - 1)) * Nn + n) * Dn];
        float a = sc[t] * rs;
        for (int dd = 0; dd < Dn; dd++) xacc[dd] += a * xr[dd];
    }
    const float* v0 = &vh0[(Pi * Nn + n) * Dn];
    const float a0 = sc[0] * rs;
    for (int d = 0; d < Dn; d++) {
        float s = a0 * v0[d];
        for (int dd = 0; dd < Dn; dd++) s += xacc[dd] * swv[dd * Dn + d];
        ipre[((size_t)(b * P + Pi) * Nn + n) * Dn + d] = s;
    }
}

// fused tail: intra_lin + inter-gather + LN1 + FF(gelu) + LN2, gated accumulate
template <int p>
__global__ void k_tail(const float* __restrict__ x, const float* __restrict__ O2,
                       const float* __restrict__ ipre,
                       const float* __restrict__ ilw, const float* __restrict__ ilb,
                       const float* __restrict__ ln1g, const float* __restrict__ ln1b,
                       const float* __restrict__ ffw1, const float* __restrict__ ffb1,
                       const float* __restrict__ ffw2, const float* __restrict__ ffb2,
                       const float* __restrict__ ln2g, const float* __restrict__ ln2b,
                       const float* __restrict__ gates, int eidx, float* __restrict__ out) {
    constexpr int P = Ln / p, Dp = Dn * p;
    const int b = blockIdx.x / Nn, n = blockIdx.x % Nn;
    const float gate = gates[b * En + eidx];
    if (gate == 0.0f) return;                  // uniform per block; reference contribution is 0
    __shared__ float s_ip[P * Dn];
    __shared__ float s_w1[Dn * DFFn], s_w2[DFFn * Dn];
    __shared__ float s_b1[DFFn], s_b2[Dn];
    __shared__ float s_g1[Dn], s_bb1[Dn], s_g2[Dn], s_bb2[Dn];
    const int l = threadIdx.x;                 // 192 threads
    for (int i = l; i < P * Dn; i += blockDim.x) {
        int Pi = i / Dn, d = i % Dn;
        s_ip[i] = ipre[((size_t)(b * P + Pi) * Nn + n) * Dn + d];
    }
    for (int i = l; i < Dn * DFFn; i += blockDim.x) s_w1[i] = ffw1[i];
    for (int i = l; i < DFFn * Dn; i += blockDim.x) s_w2[i] = ffw2[i];
    for (int i = l; i < DFFn; i += blockDim.x) s_b1[i] = ffb1[i];
    for (int i = l; i < Dn; i += blockDim.x) {
        s_b2[i] = ffb2[i];
        s_g1[i] = ln1g[i]; s_bb1[i] = ln1b[i];
        s_g2[i] = ln2g[i]; s_bb2[i] = ln2b[i];
    }
    __syncthreads();

    const float* xr = &x[((size_t)(b * Ln + l) * Nn + n) * Dn];
    const float bl = ilb[l];
    float t[Dn];
    for (int d = 0; d < Dn; d++) {
        int flat = l * (Nn * Dn) + n * Dn + d;       // reshape (n',P,Dp) -> (l,n,d)
        int n2  = flat / (P * Dp);
        int rem = flat - n2 * (P * Dp);
        int Pi  = rem / Dp;
        int k   = rem - Pi * Dp;
        float inter = O2[(((size_t)b * Nn + n2) * P + Pi) * Dp + k];
        float intra = bl;
        for (int q = 0; q < P; q++) intra += s_ip[q * Dn + d] * ilw[q * Ln + l];
        t[d] = xr[d] + intra + inter;
    }
    // LN1
    float mean = 0.f;
    for (int d = 0; d < Dn; d++) mean += t[d];
    mean *= (1.0f / Dn);
    float var = 0.f;
    for (int d = 0; d < Dn; d++) { float u = t[d] - mean; var += u * u; }
    var *= (1.0f / Dn);
    float rstd = rsqrtf(var + 1e-5f);
    float y1[Dn];
    for (int d = 0; d < Dn; d++) y1[d] = (t[d] - mean) * rstd * s_g1[d] + s_bb1[d];
    // FF
    float h[DFFn];
    for (int j = 0; j < DFFn; j++) {
        float s = s_b1[j];
        for (int d = 0; d < Dn; d++) s += y1[d] * s_w1[d * DFFn + j];
        h[j] = gelu_f(s);
    }
    for (int d = 0; d < Dn; d++) {
        float s = s_b2[d];
        for (int j = 0; j < DFFn; j++) s += h[j] * s_w2[j * Dn + d];
        t[d] = y1[d] + s;
    }
    // LN2 + gated accumulate
    mean = 0.f;
    for (int d = 0; d < Dn; d++) mean += t[d];
    mean *= (1.0f / Dn);
    var = 0.f;
    for (int d = 0; d < Dn; d++) { float u = t[d] - mean; var += u * u; }
    var *= (1.0f / Dn);
    rstd = rsqrtf(var + 1e-5f);
    float* op = &out[((size_t)(b * Ln + l) * Nn + n) * Dn];
    for (int d = 0; d < Dn; d++) {
        float y = (t[d] - mean) * rstd * s_g2[d] + s_bb2[d];
        op[d] += gate * y;
    }
}

// ---------------- host-side expert driver ----------------
template <int p>
static void run_expert(const float* x, const float* const* prm, const float* gates,
                       float* qh_c, float* kh0, float* vh0, float* ipre,
                       float* W0, float* W1, float* W2, float* W3, float* W4,
                       int e, float* out, hipStream_t stream) {
    constexpr int P = Ln / p, Dp = Dn * p;
    constexpr int R = Bn * Nn * P;
    k_pack<p><<<(OUTMAIN + 255) / 256, 256, 0, stream>>>(x, W0);
    dim3 gblk(R / 64, Dp / 64);
    // z = z0 @ emb_lin_w + emb_lin_b + w_pos
    k_gemm<<<gblk, 128, 0, stream>>>(W0, prm[8], prm[9], prm[10], W1, R, Dp, Dp, P);
    // q2/k2/v2
    k_gemm<<<gblk, 128, 0, stream>>>(W1, prm[11], prm[12], nullptr, W2, R, Dp, Dp, 1);
    k_gemm<<<gblk, 128, 0, stream>>>(W1, prm[13], prm[14], nullptr, W3, R, Dp, Dp, 1);
    k_gemm<<<gblk, 128, 0, stream>>>(W1, prm[15], prm[16], nullptr, W4, R, Dp, Dp, 1);
    k_attn<p><<<Bn * Nn * Hn * P, 128, 0, stream>>>(W2, W3, W4, W0);
    // o @ Wo + bo
    k_gemm<<<gblk, 128, 0, stream>>>(W0, prm[17], prm[18], nullptr, W1, R, Dp, Dp, 1);
    // intra attention
    k_intra_q<<<(P * Nn + 127) / 128, 128, 0, stream>>>(prm[0], prm[1], prm[2],
                                                        prm[3], prm[4], prm[5],
                                                        qh_c, kh0, vh0, P);
    k_intra<p><<<(Bn * P * Nn + 255) / 256, 256, 0, stream>>>(x, prm[4], prm[5],
                                                              qh_c, kh0, vh0, ipre);
    // fused tail
    k_tail<p><<<Bn * Nn, Ln, 0, stream>>>(x, W1, ipre, prm[6], prm[7],
                                          prm[19], prm[20], prm[21], prm[22],
                                          prm[23], prm[24], prm[25], prm[26],
                                          gates, e, out);
}

extern "C" void kernel_launch(void* const* d_in, const int* in_sizes, int n_in,
                              void* d_out, int out_size, void* d_ws, size_t ws_size,
                              hipStream_t stream) {
    (void)in_sizes; (void)n_in; (void)out_size; (void)ws_size;
    const float* x       = (const float*)d_in[0];
    const float* w_start = (const float*)d_in[1];
    const float* b_start = (const float*)d_in[2];
    const float* w_gate  = (const float*)d_in[3];
    const float* b_gate  = (const float*)d_in[4];
    const float* w_dec   = (const float*)d_in[5];
    const float* b_dec   = (const float*)d_in[6];
    float* out = (float*)d_out;

    // workspace layout (floats)
    float* ws = (float*)d_ws;
    size_t off = 0;
    float* newx  = ws + off; off += (size_t)Bn * Ln * Nn;      // 393216
    float* gbuf  = ws + off; off += (size_t)Bn * Ln;           // 6144
    float* gates = ws + off; off += (size_t)Bn * En + 32;      // 128 (+pad for alignment)
    float* qh_c  = ws + off; off += (size_t)96 * Nn * Dn;      // max-P sized
    float* kh0   = ws + off; off += (size_t)96 * Nn * Dn;
    float* vh0   = ws + off; off += (size_t)96 * Nn * Dn;
    float* ipre  = ws + off; off += (size_t)Bn * 96 * Nn * Dn; // 6291456
    float* W0 = ws + off; off += (size_t)OUTMAIN;
    float* W1 = ws + off; off += (size_t)OUTMAIN;
    float* W2 = ws + off; off += (size_t)OUTMAIN;
    float* W3 = ws + off; off += (size_t)OUTMAIN;
    float* W4 = ws + off; off += (size_t)OUTMAIN;

    // stage 0/1: init out, gating pipeline
    k_init_out<<<(OUTMAIN / 4 + 255) / 256, 256, 0, stream>>>((const float4*)x, (float4*)out);
    k_season_trend<<<Bn * Nn, Ln, 0, stream>>>(x, w_dec, b_dec, newx);
    k_g<<<(Bn * Ln + 255) / 256, 256, 0, stream>>>(newx, w_start, b_start, gbuf);
    k_gate<<<1, 32, 0, stream>>>(gbuf, w_gate, b_gate, gates, out + OUTMAIN);

    // experts
    static const int patches[4] = {8, 6, 4, 2};
    for (int e = 0; e < 4; e++) {
        const float* prm[27];
        for (int i = 0; i < 27; i++) prm[i] = (const float*)d_in[7 + e * 27 + i];
        switch (patches[e]) {
            case 8: run_expert<8>(x, prm, gates, qh_c, kh0, vh0, ipre, W0, W1, W2, W3, W4, e, out, stream); break;
            case 6: run_expert<6>(x, prm, gates, qh_c, kh0, vh0, ipre, W0, W1, W2, W3, W4, e, out, stream); break;
            case 4: run_expert<4>(x, prm, gates, qh_c, kh0, vh0, ipre, W0, W1, W2, W3, W4, e, out, stream); break;
            case 2: run_expert<2>(x, prm, gates, qh_c, kh0, vh0, ipre, W0, W1, W2, W3, W4, e, out, stream); break;
        }
    }
}
